// QCNN_ZNOTY_Shared_12756052869701
// MI455X (gfx1250) — matrix operand order repaired and checked by reading, not compiled, not run on hardware
//
#include <hip/hip_runtime.h>
#include <hip/hip_bf16.h>
#include <math.h>

// ---------------------------------------------------------------------------
// QCNN collapse:  out = MLP( x^T M_w x / ||x||^2 ),  M_w = Re(U^ D_w U), w in {3,7}
//   kernel 1: build U (256x256 complex) by evolving basis columns
//   kernel 2: build M (512x256, bf16)  -- L2-resident afterwards
//   kernel 3: fused bf16-WMMA GEMM + quadratic form + MLP (HBM-roofline bound)
// ---------------------------------------------------------------------------

typedef __attribute__((ext_vector_type(16))) __bf16 v16bf;
typedef __attribute__((ext_vector_type(8)))  float  v8f;

union Frag16 {
    v16bf v;
    unsigned int u[8];
    uint4 q[2];
};

__device__ __forceinline__ unsigned int f2bf(float f) {
    unsigned int b = __float_as_uint(f);
    b += 0x7FFFu + ((b >> 16) & 1u);     // round-to-nearest-even
    return b >> 16;
}
__device__ __forceinline__ unsigned int pack_bf2(float lo, float hi) {
    return f2bf(lo) | (f2bf(hi) << 16);
}

// ---------------------------------------------------------------------------
// Kernel 1: evolve each basis column through the circuit -> U (256x256 cplx)
// One block per column, 128 threads, state lives in 2KB LDS.
// Qubit w <-> bit (7 - w) of the flat basis index.
// ---------------------------------------------------------------------------
__global__ __launch_bounds__(128)
void qcnn_build_u(const float* __restrict__ conv_rz, const float* __restrict__ conv_ry,
                  const float* __restrict__ conv_ry2, const float* __restrict__ pool,
                  float* __restrict__ Ur, float* __restrict__ Ui) {
    __shared__ float sr[256];
    __shared__ float si[256];
    const int tid = threadIdx.x;
    const int col = blockIdx.x;
    for (int k = tid; k < 256; k += 128) {
        sr[k] = (k == col) ? 1.0f : 0.0f;
        si[k] = 0.0f;
    }

    auto ap1 = [&](int w, float u00r, float u00i, float u01r, float u01i,
                          float u10r, float u10i, float u11r, float u11i) {
        __syncthreads();
        const int b  = 7 - w;
        const int p  = tid;                               // 128 disjoint pairs
        const int i0 = ((p >> b) << (b + 1)) | (p & ((1 << b) - 1));
        const int i1 = i0 | (1 << b);
        const float a0r = sr[i0], a0i = si[i0];
        const float a1r = sr[i1], a1i = si[i1];
        sr[i0] = u00r * a0r - u00i * a0i + u01r * a1r - u01i * a1i;
        si[i0] = u00r * a0i + u00i * a0r + u01r * a1i + u01i * a1r;
        sr[i1] = u10r * a0r - u10i * a0i + u11r * a1r - u11i * a1i;
        si[i1] = u10r * a0i + u10i * a0r + u11r * a1i + u11i * a1r;
    };
    auto cx = [&](int wc, int wt) {                       // control wc, target wt
        __syncthreads();
        if (tid < 64) {
            const int bc = 7 - wc, bt = 7 - wt;
            const int bl = bc < bt ? bc : bt;
            const int bh = bc < bt ? bt : bc;
            int i = tid;
            i = ((i >> bl) << (bl + 1)) | (i & ((1 << bl) - 1));
            i = ((i >> bh) << (bh + 1)) | (i & ((1 << bh) - 1));
            const int i0 = i | (1 << bc);
            const int i1 = i0 | (1 << bt);
            float tr = sr[i0], ti = si[i0];
            sr[i0] = sr[i1]; si[i0] = si[i1];
            sr[i1] = tr;     si[i1] = ti;
        }
    };
    auto rz = [&](int w, float t) {
        const float c = __cosf(0.5f * t), s = __sinf(0.5f * t);
        ap1(w, c, -s, 0.f, 0.f, 0.f, 0.f, c, s);
    };
    auto ry = [&](int w, float t) {
        const float c = __cosf(0.5f * t), s = __sinf(0.5f * t);
        ap1(w, c, 0.f, -s, 0.f, s, 0.f, c, 0.f);
    };
    auto conv = [&](int qa, int qb, float trz, float tryy, float try2) {
        const float PI_ = 3.14159265358979323846f;
        rz(qb, -0.5f * PI_);
        cx(qb, qa);
        rz(qa, trz);
        ry(qb, tryy);
        cx(qa, qb);
        ry(qb, try2);
        cx(qb, qa);
        rz(qa, 0.5f * PI_);
    };
    auto u3g = [&](int w, float t, float p, float l) {
        const float c = __cosf(0.5f * t), s = __sinf(0.5f * t);
        const float cl = __cosf(l),  sl = __sinf(l);
        const float cp = __cosf(p),  sp = __sinf(p);
        const float cq = __cosf(p + l), sq = __sinf(p + l);
        ap1(w, c, 0.f, -cl * s, -sl * s, cp * s, sp * s, cq * c, sq * c);
    };

    {   // layer 0
        const float a = conv_rz[0], b = conv_ry[0], c = conv_ry2[0];
        conv(0, 1, a, b, c); conv(2, 3, a, b, c); conv(4, 5, a, b, c); conv(6, 7, a, b, c);
        conv(1, 2, a, b, c); conv(3, 4, a, b, c); conv(5, 6, a, b, c);
        const float t = pool[0], p = pool[1], l = pool[2];
        u3g(1, t, p, l); u3g(3, t, p, l); u3g(5, t, p, l); u3g(7, t, p, l);
    }
    {   // layer 1
        const float a = conv_rz[1], b = conv_ry[1], c = conv_ry2[1];
        conv(1, 3, a, b, c); conv(5, 7, a, b, c); conv(3, 5, a, b, c);
        const float t = pool[3], p = pool[4], l = pool[5];
        u3g(3, t, p, l); u3g(7, t, p, l);
    }

    __syncthreads();
    for (int k = tid; k < 256; k += 128) {
        Ur[k * 256 + col] = sr[k];
        Ui[k * 256 + col] = si[k];
    }
}

// ---------------------------------------------------------------------------
// Kernel 2: M[n][j] = sum_k d_k * (Ur[k][i]Ur[k][j] + Ui[k][i]Ui[k][j])
// n = w*256 + i ; wire 3 -> bit 4, wire 7 -> bit 0. Output bf16 row-major.
// ---------------------------------------------------------------------------
__global__ __launch_bounds__(256)
void qcnn_build_m(const float* __restrict__ Ur, const float* __restrict__ Ui,
                  unsigned short* __restrict__ Mbf) {
    const int n    = blockIdx.x;        // 0..511
    const int j    = threadIdx.x;       // 0..255
    const int wsel = n >> 8;
    const int i    = n & 255;
    const int bit  = wsel ? 0 : 4;      // 7 - wire
    float acc = 0.0f;
    for (int k = 0; k < 256; ++k) {
        const float d  = ((k >> bit) & 1) ? -1.0f : 1.0f;
        const float ri = Ur[k * 256 + i];   // block-uniform -> scalar load
        const float ii = Ui[k * 256 + i];
        acc += d * (ri * Ur[k * 256 + j] + ii * Ui[k * 256 + j]);
    }
    Mbf[n * 256 + j] = (unsigned short)f2bf(acc);
}

// ---------------------------------------------------------------------------
// Kernel 3: fused  y = M x  ->  ev_w = <x,y_w>/||x||^2  ->  MLP.
// 256 threads (8 waves) per block; each wave owns TWO 16-row M-tiles (32 rows)
// -> 256 batch rows / block, 256 blocks.  Per N-tile, the 8KB B slab of M is
// staged once into double-buffered LDS and shared by all 8 waves:
//   M L2 traffic = 256 blocks * 256KB = 64MB  ==  compulsory x HBM read.
// A-fragments built from global f32 (each element read once) + RNE bf16 pack.
// ---------------------------------------------------------------------------
__global__ __launch_bounds__(256)
void qcnn_main(const float* __restrict__ x, const uint4* __restrict__ Mq4,
               const float* __restrict__ W1, const float* __restrict__ b1,
               const float* __restrict__ W2, const float* __restrict__ b2,
               float* __restrict__ out) {
    __shared__ unsigned int Bbuf[2][2048];   // 2 x 8KB B slabs (16 rows x 256 bf16)
    __shared__ float lds_norm[256];
    __shared__ float lds_ev[256][2];

    const int tid  = threadIdx.x;
    const int row0 = blockIdx.x * 256;
    const int wave = tid >> 5;
    const int lane = tid & 31;
    const int lrow = lane & 15;          // A row / B col / C col within tile
    const int lhi  = lane >> 4;          // half-wave selector
    const int kbase = lhi * 8;

    lds_norm[tid]  = 0.0f;
    lds_ev[tid][0] = 0.0f;
    lds_ev[tid][1] = 0.0f;
    __syncthreads();

    // ---- build A fragments for both tiles directly from global f32 x,
    //      accumulating row-norm partial sums along the way.
    Frag16 a[2][8];
    #pragma unroll
    for (int t = 0; t < 2; ++t) {
        const int lr = wave * 32 + t * 16 + lrow;            // local row
        const float* xr = x + (size_t)(row0 + lr) * 256;
        float nrm = 0.0f;
        #pragma unroll
        for (int kk = 0; kk < 8; ++kk) {
            const float4 p0 = *(const float4*)(xr + kk * 32 + kbase);
            const float4 p1 = *(const float4*)(xr + kk * 32 + kbase + 4);
            const float4 p2 = *(const float4*)(xr + kk * 32 + kbase + 16);
            const float4 p3 = *(const float4*)(xr + kk * 32 + kbase + 20);
            a[t][kk].u[0] = pack_bf2(p0.x, p0.y);
            a[t][kk].u[1] = pack_bf2(p0.z, p0.w);
            a[t][kk].u[2] = pack_bf2(p1.x, p1.y);
            a[t][kk].u[3] = pack_bf2(p1.z, p1.w);
            a[t][kk].u[4] = pack_bf2(p2.x, p2.y);
            a[t][kk].u[5] = pack_bf2(p2.z, p2.w);
            a[t][kk].u[6] = pack_bf2(p3.x, p3.y);
            a[t][kk].u[7] = pack_bf2(p3.z, p3.w);
            nrm += p0.x*p0.x + p0.y*p0.y + p0.z*p0.z + p0.w*p0.w
                 + p1.x*p1.x + p1.y*p1.y + p1.z*p1.z + p1.w*p1.w
                 + p2.x*p2.x + p2.y*p2.y + p2.z*p2.z + p2.w*p2.w
                 + p3.x*p3.x + p3.y*p3.y + p3.z*p3.z + p3.w*p3.w;
        }
        atomicAdd(&lds_norm[lr], nrm);      // two lanes (L, L+16) cover the row
    }

    // ---- prologue: stage B slab for nt = 0 (coalesced 32B per thread)
    {
        uint4 s0 = Mq4[tid * 2 + 0];
        uint4 s1 = Mq4[tid * 2 + 1];
        ((uint4*)&Bbuf[0][0])[tid * 2 + 0] = s0;
        ((uint4*)&Bbuf[0][0])[tid * 2 + 1] = s1;
    }
    __syncthreads();

    float ev0[2][8] = {};
    float ev1[2][8] = {};

    for (int nt = 0; nt < 32; ++nt) {
        // issue next slab's global loads early (latency overlaps compute)
        const int nn = (nt + 1 < 32) ? (nt + 1) : 31;
        uint4 s0 = Mq4[nn * 512 + tid * 2 + 0];
        uint4 s1 = Mq4[nn * 512 + tid * 2 + 1];

        // GEMM: 8 K-steps x 2 M-tiles from the shared LDS B slab
        const uint4* bp = (const uint4*)&Bbuf[nt & 1][0];   // runtime cast, no static init
        v8f c0 = {0.f,0.f,0.f,0.f,0.f,0.f,0.f,0.f};
        v8f c1 = {0.f,0.f,0.f,0.f,0.f,0.f,0.f,0.f};
        #pragma unroll
        for (int kk = 0; kk < 8; ++kk) {
            Frag16 bfr;                       // B 32x16 fragment (ISA 7.12.2)
            const int bidx = lrow * 32 + kk * 4 + lhi;
            bfr.q[0] = bp[bidx];
            bfr.q[1] = bp[bidx + 2];
            c0 = __builtin_amdgcn_wmma_f32_16x16x32_bf16(
                     false, a[0][kk].v, false, bfr.v, (short)0, c0, false, false);
            c1 = __builtin_amdgcn_wmma_f32_16x16x32_bf16(
                     false, a[1][kk].v, false, bfr.v, (short)0, c1, false, false);
        }

        // fold C tiles into <x, y>; C(m,n): m = r + 8*lhi, n = lrow
        const int j = (nt & 15) * 16 + lrow;            // column index into x
        #pragma unroll
        for (int r = 0; r < 8; ++r) {
            const int m0 = wave * 32 + r + 8 * lhi;     // tile 0 local row
            const int m1 = m0 + 16;                     // tile 1 local row
            const float xa = x[(size_t)(row0 + m0) * 256 + j];
            const float xb = x[(size_t)(row0 + m1) * 256 + j];
            const float ca = c0[r] * xa;
            const float cb = c1[r] * xb;
            if (nt < 16) { ev0[0][r] += ca; ev0[1][r] += cb; }
            else         { ev1[0][r] += ca; ev1[1][r] += cb; }
        }

        __syncthreads();                                // all waves done reading buf[(nt+1)&1]
        ((uint4*)&Bbuf[(nt + 1) & 1][0])[tid * 2 + 0] = s0;
        ((uint4*)&Bbuf[(nt + 1) & 1][0])[tid * 2 + 1] = s1;
        __syncthreads();                                // staged slab visible
    }

    #pragma unroll
    for (int t = 0; t < 2; ++t) {
        #pragma unroll
        for (int r = 0; r < 8; ++r) {
            const int m = wave * 32 + t * 16 + r + 8 * lhi;
            atomicAdd(&lds_ev[m][0], ev0[t][r]);
            atomicAdd(&lds_ev[m][1], ev1[t][r]);
        }
    }
    __syncthreads();

    // ---- per-row MLP: ev(2) -> tanh(10) -> sigmoid(1)
    {
        const float inv = 1.0f / lds_norm[tid];
        const float e0 = lds_ev[tid][0] * inv;
        const float e1 = lds_ev[tid][1] * inv;
        float acc = b2[0];
        #pragma unroll
        for (int jj = 0; jj < 10; ++jj) {
            const float h = tanhf(e0 * W1[jj] + e1 * W1[10 + jj] + b1[jj]);
            acc += h * W2[jj];
        }
        out[row0 + tid] = 1.0f / (1.0f + __expf(-acc));
    }
}

// ---------------------------------------------------------------------------
extern "C" void kernel_launch(void* const* d_in, const int* in_sizes, int n_in,
                              void* d_out, int out_size, void* d_ws, size_t ws_size,
                              hipStream_t stream) {
    const float* x        = (const float*)d_in[0];
    const float* conv_rz  = (const float*)d_in[1];
    const float* conv_ry  = (const float*)d_in[2];
    const float* conv_ry2 = (const float*)d_in[3];
    const float* pool     = (const float*)d_in[4];
    const float* W1       = (const float*)d_in[5];
    const float* b1       = (const float*)d_in[6];
    const float* W2       = (const float*)d_in[7];
    const float* b2       = (const float*)d_in[8];
    float* out = (float*)d_out;

    char* ws = (char*)d_ws;
    unsigned short* Mbf = (unsigned short*)ws;                        // 512*256*2 = 256KB
    float* Ur = (float*)(ws + 512 * 256 * 2);                         // 256KB
    float* Ui = (float*)(ws + 512 * 256 * 2 + 256 * 256 * 4);         // 256KB

    const int B = in_sizes[0] / 256;                                  // 65536

    qcnn_build_u<<<256, 128, 0, stream>>>(conv_rz, conv_ry, conv_ry2, pool, Ur, Ui);
    qcnn_build_m<<<512, 256, 0, stream>>>(Ur, Ui, Mbf);
    qcnn_main<<<B / 256, 256, 0, stream>>>(x, (const uint4*)Mbf,
                                           W1, b1, W2, b2, out);
}
